// MHAttention_36807869727215
// MI455X (gfx1250) — matrix operand order repaired and checked by reading, not compiled, not run on hardware
//
#include <hip/hip_runtime.h>

typedef _Float16 half_t;
typedef __attribute__((ext_vector_type(16))) _Float16 v16h;
typedef __attribute__((ext_vector_type(8)))  _Float16 v8h;
typedef __attribute__((ext_vector_type(4)))  _Float16 v4h;
typedef __attribute__((ext_vector_type(8)))  float    v8f;
typedef __attribute__((ext_vector_type(4)))  float    v4f;

#define DIMN   1024
#define HEADS  16
#define HD     64
#define BATCH  4
#define SEQ    2048

// ---------------------------------------------------------------------------
// Async global -> LDS copy (CDNA5 GLOBAL_LOAD_ASYNC_TO_LDS_B128, ASYNCcnt).
// LDS byte address = low 32 bits of the generic pointer (aperture in high bits).
// ---------------------------------------------------------------------------
__device__ __forceinline__ void async_ld_b128(void* lds_dst, const void* gsrc) {
  unsigned int laddr = (unsigned int)(unsigned long long)(uintptr_t)lds_dst;
  unsigned long long gaddr = (unsigned long long)(uintptr_t)gsrc;
  asm volatile("global_load_async_to_lds_b128 %0, %1, off"
               :: "v"(laddr), "v"(gaddr)
               : "memory");
}
__device__ __forceinline__ void wait_async0() {
  asm volatile("s_wait_asynccnt 0" ::: "memory");
}

// ---------------------------------------------------------------------------
// WMMA helpers (CDNA5 wave32, V_WMMA_F32_16X16X32_F16)
// ---------------------------------------------------------------------------
__device__ __forceinline__ v16h make_v16(v8h lo, v8h hi) {
  union { v16h v; v8h h[2]; } u;
  u.h[0] = lo; u.h[1] = hi;
  return u.v;
}

// A fragment 16x32 f16 from LDS stored [row][k] with row stride ld (halves).
__device__ __forceinline__ v16h ldsA(const half_t* p, int ld, int mBase, int kBase, int lane) {
  int row = mBase + (lane & 15);
  int k0  = kBase + ((lane >> 4) << 3);
  const half_t* q = p + row * ld + k0;
  v8h lo = *(const v8h*)(q);
  v8h hi = *(const v8h*)(q + 16);
  return make_v16(lo, hi);
}

// B fragment 32x16 f16 from LDS stored TRANSPOSED [n][k] with stride ld (halves).
__device__ __forceinline__ v16h ldsB(const half_t* p, int ld, int nBase, int kBase, int lane) {
  int col = nBase + (lane & 15);
  int k0  = kBase + ((lane >> 4) << 3);
  const half_t* q = p + col * ld + k0;
  v8h lo = *(const v8h*)(q);
  v8h hi = *(const v8h*)(q + 16);
  return make_v16(lo, hi);
}

__device__ __forceinline__ v8f wmma_f16(v16h a, v16h b, v8f c) {
  return __builtin_amdgcn_wmma_f32_16x16x32_f16(false, a, false, b, (short)0, c, false, false);
}

// ---------------------------------------------------------------------------
// Weight conversion fp32 -> f16
// ---------------------------------------------------------------------------
__global__ __launch_bounds__(256) void cvt_f32_f16(const float* __restrict__ src,
                                                   half_t* __restrict__ dst, int n) {
  int i = (blockIdx.x * 256 + threadIdx.x) * 4;
  if (i + 3 < n) {
    v4f f = *(const v4f*)(src + i);
    v4h h;
    h[0] = (half_t)f[0]; h[1] = (half_t)f[1];
    h[2] = (half_t)f[2]; h[3] = (half_t)f[3];
    *(v4h*)(dst + i) = h;
  }
}

// ---------------------------------------------------------------------------
// Tiled WMMA GEMM: C[M,N] = A[M,K] * B[K,N]
// Block tile 128x128, K-step 32, 256 threads = 8 waves, wave tile 32x64.
// ---------------------------------------------------------------------------
template <typename AT, typename OT>
__global__ __launch_bounds__(256)
void gemm_wmma(const AT* __restrict__ A, const half_t* __restrict__ B,
               OT* __restrict__ C, int M, int N, int K) {
  __shared__ __align__(16) half_t As[128 * 40];   // [m][k], stride 40
  __shared__ __align__(16) half_t Bs[128 * 40];   // transposed: [n][k], stride 40

  const int tid  = threadIdx.x;
  const int lane = tid & 31;
  const int wid  = tid >> 5;
  const int wm   = (wid & 3) * 32;
  const int wn   = (wid >> 2) * 64;
  const int mBase = blockIdx.y * 128;
  const int nBase = blockIdx.x * 128;

  v8f zf = {};
  v8f acc[2][4];
  #pragma unroll
  for (int i = 0; i < 2; ++i)
    #pragma unroll
    for (int j = 0; j < 4; ++j) acc[i][j] = zf;

  for (int kt = 0; kt < K; kt += 32) {
    __syncthreads();
    // ---- stage A tile (128 x 32) into LDS as f16 ----
    if constexpr (sizeof(AT) == 4) {
      #pragma unroll
      for (int it = 0; it < 4; ++it) {
        int idx = it * 1024 + tid * 4;
        int r = idx >> 5, c = idx & 31;
        v4f f = *(const v4f*)(A + (size_t)(mBase + r) * K + kt + c);
        half_t* d = As + r * 40 + c;
        d[0] = (half_t)f[0]; d[1] = (half_t)f[1];
        d[2] = (half_t)f[2]; d[3] = (half_t)f[3];
      }
    } else {
      // pure copy: use async DMA path (ASYNCcnt), 16B per lane per iter
      #pragma unroll
      for (int it = 0; it < 2; ++it) {
        int idx = it * 2048 + tid * 8;
        int r = idx >> 5, c = idx & 31;
        async_ld_b128(As + r * 40 + c, A + (size_t)(mBase + r) * K + kt + c);
      }
    }
    // ---- stage B tile (32 x 128) transposed into LDS [n][k] ----
    #pragma unroll
    for (int it = 0; it < 2; ++it) {
      int idx = it * 2048 + tid * 8;
      int kr = idx >> 7, c = idx & 127;
      v8h h = *(const v8h*)(B + (size_t)(kt + kr) * N + nBase + c);
      #pragma unroll
      for (int j = 0; j < 8; ++j) Bs[(c + j) * 40 + kr] = h[j];
    }
    if (kt + 32 < K) {
      __builtin_prefetch(A + (size_t)(mBase + (tid >> 1)) * K + kt + 32, 0, 0);
      __builtin_prefetch(B + (size_t)(kt + 32 + (tid & 31)) * N + nBase, 0, 0);
    }
    if constexpr (sizeof(AT) == 2) wait_async0();
    __syncthreads();

    // ---- compute: 2 (M-frags) x 4 (N-frags) WMMA ----
    v16h a0 = ldsA(As, 40, wm, 0, lane);
    v16h a1 = ldsA(As, 40, wm + 16, 0, lane);
    #pragma unroll
    for (int fn = 0; fn < 4; ++fn) {
      v16h b = ldsB(Bs, 40, wn + fn * 16, 0, lane);
      acc[0][fn] = wmma_f16(a0, b, acc[0][fn]);
      acc[1][fn] = wmma_f16(a1, b, acc[1][fn]);
    }
  }

  // ---- epilogue ----
  const int hi = lane >> 4, ln = lane & 15;
  #pragma unroll
  for (int fm = 0; fm < 2; ++fm)
    #pragma unroll
    for (int fn = 0; fn < 4; ++fn)
      #pragma unroll
      for (int r = 0; r < 8; ++r) {
        int row = mBase + wm + fm * 16 + r + hi * 8;
        int col = nBase + wn + fn * 16 + ln;
        C[(size_t)row * N + col] = (OT)acc[fm][fn][r];
      }
}

// ---------------------------------------------------------------------------
// Flash attention: one block = (b,h) pair x 64 query rows; 4 waves x 16 rows.
// ---------------------------------------------------------------------------
__global__ __launch_bounds__(128)
void attn_kernel(const half_t* __restrict__ Qh, const half_t* __restrict__ Kh,
                 const half_t* __restrict__ Vh, const float* __restrict__ mask,
                 half_t* __restrict__ att) {
  __shared__ __align__(16) half_t Qs[64 * 72];  // [qrow][d]
  __shared__ __align__(16) half_t Ks[64 * 64];  // [key][d]   (B-layout for QK^T)
  __shared__ __align__(16) half_t Vt[64 * 72];  // [d][key]   (B-layout for P*V)
  __shared__ __align__(16) half_t Ps[64 * 72];  // [qrow][key]
  __shared__ float mb[64];

  const int tid = threadIdx.x, lane = tid & 31, wid = tid >> 5;
  const int hi = lane >> 4, ln = lane & 15;
  const int bh = blockIdx.y, b = bh >> 4, h = bh & 15;
  const int qBase = blockIdx.x * 64;
  const size_t headOff = (size_t)h * HD;

  // ---- load Q tile once (async DMA copy) ----
  #pragma unroll
  for (int it = 0; it < 4; ++it) {
    int idx = it * 1024 + tid * 8;
    int r = idx >> 6, c = idx & 63;
    async_ld_b128(Qs + r * 72 + c,
                  Qh + (size_t)(b * SEQ + qBase + r) * DIMN + headOff + c);
  }
  wait_async0();
  __syncthreads();
  v16h aq0 = ldsA(Qs, 72, wid * 16, 0, lane);
  v16h aq1 = ldsA(Qs, 72, wid * 16, 32, lane);

  v8f zf = {};
  v8f o[4];
  #pragma unroll
  for (int i = 0; i < 4; ++i) o[i] = zf;
  float mrun[8], lrun[8];
  #pragma unroll
  for (int r = 0; r < 8; ++r) { mrun[r] = -3.0e38f; lrun[r] = 0.f; }

  for (int kb = 0; kb < SEQ; kb += 64) {
    __syncthreads();
    // ---- stage K (async copy) and V (manual transpose) tiles ----
    #pragma unroll
    for (int it = 0; it < 4; ++it) {
      int idx = it * 1024 + tid * 8;
      int r = idx >> 6, c = idx & 63;
      async_ld_b128(Ks + r * 64 + c,
                    Kh + (size_t)(b * SEQ + kb + r) * DIMN + headOff + c);
      v8h vx = *(const v8h*)(Vh + (size_t)(b * SEQ + kb + r) * DIMN + headOff + c);
      #pragma unroll
      for (int j = 0; j < 8; ++j) Vt[(c + j) * 72 + r] = vx[j];
    }
    if (tid < 64) mb[tid] = -1.0e9f * mask[(size_t)b * SEQ + kb + tid];
    wait_async0();
    __syncthreads();

    // ---- S = Q K^T (contraction over d=64, two WMMA k-steps) ----
    v8f s[4];
    #pragma unroll
    for (int fn = 0; fn < 4; ++fn) {
      v16h b0 = ldsB(Ks, 64, fn * 16, 0, lane);
      v16h b1 = ldsB(Ks, 64, fn * 16, 32, lane);
      v8f a = zf;
      a = wmma_f16(aq0, b0, a);
      a = wmma_f16(aq1, b1, a);
      s[fn] = a;
    }
    // ---- scale + padding-mask bias ----
    #pragma unroll
    for (int fn = 0; fn < 4; ++fn) {
      float bias = mb[fn * 16 + ln];
      #pragma unroll
      for (int r = 0; r < 8; ++r) s[fn][r] = s[fn][r] * 0.125f + bias;
    }
    // ---- online softmax (rows live in 16-lane halves; xor<=8 stays inside) ----
    #pragma unroll
    for (int r = 0; r < 8; ++r) {
      float m = fmaxf(fmaxf(s[0][r], s[1][r]), fmaxf(s[2][r], s[3][r]));
      m = fmaxf(m, __shfl_xor(m, 1));
      m = fmaxf(m, __shfl_xor(m, 2));
      m = fmaxf(m, __shfl_xor(m, 4));
      m = fmaxf(m, __shfl_xor(m, 8));
      float mnew  = fmaxf(mrun[r], m);
      float alpha = __expf(mrun[r] - mnew);
      float rs = 0.f;
      #pragma unroll
      for (int fn = 0; fn < 4; ++fn) {
        float p = __expf(s[fn][r] - mnew);
        s[fn][r] = p;
        rs += p;
      }
      rs += __shfl_xor(rs, 1);
      rs += __shfl_xor(rs, 2);
      rs += __shfl_xor(rs, 4);
      rs += __shfl_xor(rs, 8);
      lrun[r] = lrun[r] * alpha + rs;
      mrun[r] = mnew;
      #pragma unroll
      for (int fn = 0; fn < 4; ++fn) o[fn][r] *= alpha;
    }
    // ---- P: C-layout -> LDS -> A-layout (wave-private rows; ds in-order) ----
    #pragma unroll
    for (int fn = 0; fn < 4; ++fn)
      #pragma unroll
      for (int r = 0; r < 8; ++r)
        Ps[(wid * 16 + hi * 8 + r) * 72 + fn * 16 + ln] = (half_t)s[fn][r];
    asm volatile("s_wait_dscnt 0" ::: "memory");

    v16h ap0 = ldsA(Ps, 72, wid * 16, 0, lane);
    v16h ap1 = ldsA(Ps, 72, wid * 16, 32, lane);
    #pragma unroll
    for (int fn = 0; fn < 4; ++fn) {
      v16h bv0 = ldsB(Vt, 72, fn * 16, 0, lane);
      v16h bv1 = ldsB(Vt, 72, fn * 16, 32, lane);
      o[fn] = wmma_f16(ap0, bv0, o[fn]);
      o[fn] = wmma_f16(ap1, bv1, o[fn]);
    }
  }

  // ---- epilogue: divide by l, store f16 att ----
  #pragma unroll
  for (int fn = 0; fn < 4; ++fn)
    #pragma unroll
    for (int r = 0; r < 8; ++r) {
      int row = qBase + wid * 16 + hi * 8 + r;
      float val = o[fn][r] / lrun[r];
      att[(size_t)(b * SEQ + row) * DIMN + headOff + fn * 16 + ln] = (half_t)val;
    }
}

// ---------------------------------------------------------------------------
// Host launcher
// ---------------------------------------------------------------------------
extern "C" void kernel_launch(void* const* d_in, const int* in_sizes, int n_in,
                              void* d_out, int out_size, void* d_ws, size_t ws_size,
                              hipStream_t stream) {
  (void)in_sizes; (void)n_in; (void)out_size; (void)ws_size;
  const float* q    = (const float*)d_in[0];
  const float* k    = (const float*)d_in[1];
  const float* v    = (const float*)d_in[2];
  const float* mask = (const float*)d_in[3];
  const float* Wq   = (const float*)d_in[4];
  const float* Wk   = (const float*)d_in[5];
  const float* Wv   = (const float*)d_in[6];
  const float* Wo   = (const float*)d_in[7];
  float* out = (float*)d_out;

  char* ws = (char*)d_ws;
  const size_t wbytes = (size_t)DIMN * DIMN * sizeof(half_t);        // 2 MB
  const size_t xbytes = (size_t)BATCH * SEQ * DIMN * sizeof(half_t); // 16 MB
  half_t* Wq16 = (half_t*)(ws + 0 * wbytes);
  half_t* Wk16 = (half_t*)(ws + 1 * wbytes);
  half_t* Wv16 = (half_t*)(ws + 2 * wbytes);
  half_t* Wo16 = (half_t*)(ws + 3 * wbytes);
  half_t* Qh   = (half_t*)(ws + 4 * wbytes);
  half_t* Kh   = (half_t*)(ws + 4 * wbytes + 1 * xbytes);
  half_t* Vh   = (half_t*)(ws + 4 * wbytes + 2 * xbytes);
  half_t* att  = (half_t*)(ws + 4 * wbytes + 3 * xbytes);

  const int wn = DIMN * DIMN;
  const int cvtBlocks = wn / (256 * 4);
  cvt_f32_f16<<<cvtBlocks, 256, 0, stream>>>(Wq, Wq16, wn);
  cvt_f32_f16<<<cvtBlocks, 256, 0, stream>>>(Wk, Wk16, wn);
  cvt_f32_f16<<<cvtBlocks, 256, 0, stream>>>(Wv, Wv16, wn);
  cvt_f32_f16<<<cvtBlocks, 256, 0, stream>>>(Wo, Wo16, wn);

  const int M = BATCH * SEQ;
  dim3 gg(DIMN / 128, M / 128);
  gemm_wmma<float, half_t><<<gg, 256, 0, stream>>>(q, Wq16, Qh, M, DIMN, DIMN);
  gemm_wmma<float, half_t><<<gg, 256, 0, stream>>>(k, Wk16, Kh, M, DIMN, DIMN);
  gemm_wmma<float, half_t><<<gg, 256, 0, stream>>>(v, Wv16, Vh, M, DIMN, DIMN);

  dim3 ga(SEQ / 64, BATCH * HEADS);
  attn_kernel<<<ga, 128, 0, stream>>>(Qh, Kh, Vh, mask, att);

  gemm_wmma<half_t, float><<<gg, 256, 0, stream>>>(att, Wo16, out, M, DIMN, DIMN);
}
